// InfiniteTransformer_75866302316747
// MI455X (gfx1250) — matrix operand order repaired and checked by reading, not compiled, not run on hardware
//
#include <hip/hip_runtime.h>

// ---------------------------------------------------------------------------
// Problem constants (from reference)
// ---------------------------------------------------------------------------
#define B_    4
#define S_    4096
#define DIM_  1024
#define H_    8
#define DK_   64
#define DV_   64
#define SEG_  512
#define NSEG_ 8
#define EXP_  4
#define BS_   (B_ * S_)          // 16384 token rows
#define HID_  (EXP_ * DIM_)      // 4096

typedef __attribute__((ext_vector_type(16))) __bf16 v16bf;
typedef __attribute__((ext_vector_type(8)))  float  v8f;

#define WMMA_BF16(a, b, c) \
  __builtin_amdgcn_wmma_f32_16x16x32_bf16(false, (a), false, (b), (short)0, (c), false, false)

// ---------------------------------------------------------------------------
// CDNA5 async global->LDS copy (ASYNCcnt path), with safe fallback.
// ROCm clang signature: (v4i addrspace(1)*, v4i addrspace(3)*, imm, cpol)
// where v4i = int __attribute__((vector_size(16))).
// ---------------------------------------------------------------------------
#if __has_builtin(__builtin_amdgcn_global_load_async_to_lds_b128) && \
    __has_builtin(__builtin_amdgcn_s_wait_asynccnt)
#define HAVE_ASYNC_LDS 1
#else
#define HAVE_ASYNC_LDS 0
#endif

typedef int v4i_vs __attribute__((vector_size(16)));

__device__ __forceinline__ void copy16_g2l(const void* g, void* l) {
#if HAVE_ASYNC_LDS
  __builtin_amdgcn_global_load_async_to_lds_b128(
      (__attribute__((address_space(1))) v4i_vs*)(g),
      (__attribute__((address_space(3))) v4i_vs*)(l), 0, 0);
#else
  *(uint4*)l = *(const uint4*)g;
#endif
}

__device__ __forceinline__ void wait_async_copies() {
#if HAVE_ASYNC_LDS
  __builtin_amdgcn_s_wait_asynccnt(0);
#endif
}

__device__ __forceinline__ v8f zero_v8f() {
  v8f z = {0.f, 0.f, 0.f, 0.f, 0.f, 0.f, 0.f, 0.f};
  return z;
}

__device__ __forceinline__ float sigma_elu1(float x) {
  // elu(x)+1 : x>0 -> x+1 ; else exp(x)
  return x > 0.f ? x + 1.f : __expf(x);
}

// ---------------------------------------------------------------------------
// WMMA fragment loaders (wave32; A/B use this code's contiguous k order, not the ISA table order)
//   A (16x32, bf16): lane&15 = M row; lanes<16 hold K 0..15,
//                    lanes>=16 hold K 16..31.
//   B (32x16, bf16): lane&15 = N col; lanes<16 hold K 0..15, lanes>=16 K 16..31.
//   C/D (16x16 f32): lane&15 = N col; VGPR r = row r (lanes<16) / r+8 (lanes>=16).
// ---------------------------------------------------------------------------
__device__ __forceinline__ v16bf frag_a_kcontig(const __bf16* p, int ld) {
  const int lane = threadIdx.x & 31;
  const __bf16* q = p + (size_t)(lane & 15) * ld + ((lane >> 4) * 16);
  v16bf r;
#pragma unroll
  for (int i = 0; i < 8; ++i) r[i] = q[i];
#pragma unroll
  for (int i = 0; i < 8; ++i) r[8 + i] = q[8 + i];
  return r;
}

__device__ __forceinline__ v16bf frag_a_sigma(const __bf16* p, int ld) {
  const int lane = threadIdx.x & 31;
  const __bf16* q = p + (size_t)(lane & 15) * ld + ((lane >> 4) * 16);
  v16bf r;
#pragma unroll
  for (int i = 0; i < 8; ++i) r[i] = (__bf16)sigma_elu1((float)q[i]);
#pragma unroll
  for (int i = 0; i < 8; ++i) r[8 + i] = (__bf16)sigma_elu1((float)q[8 + i]);
  return r;
}

// sigma(K)^T fragment: A[m=feature][k=s] = elu(K[sbase+s][fbase+m])+1
__device__ __forceinline__ v16bf frag_aT_sigma(const __bf16* kmat, int fbase, int sbase, int ld) {
  const int lane = threadIdx.x & 31;
  const int f  = fbase + (lane & 15);
  const int kb = (lane >> 4) * 16;
  v16bf r;
#pragma unroll
  for (int i = 0; i < 8; ++i)
    r[i] = (__bf16)sigma_elu1((float)kmat[(size_t)(sbase + kb + i) * ld + f]);
#pragma unroll
  for (int i = 0; i < 8; ++i)
    r[8 + i] = (__bf16)sigma_elu1((float)kmat[(size_t)(sbase + 8 + kb + i) * ld + f]);
  return r;
}

// B fragment where source row n holds contiguous K elements (stride ld between n's)
__device__ __forceinline__ v16bf frag_b_kcontig(const __bf16* p, int ld) {
  const int lane = threadIdx.x & 31;
  const __bf16* q = p + (size_t)(lane & 15) * ld + ((lane >> 4) * 16);
  v16bf r;
#pragma unroll
  for (int i = 0; i < 16; ++i) r[i] = q[i];
  return r;
}

// B fragment from row-major (K x N) source: per-lane strided gather
__device__ __forceinline__ v16bf frag_b_strided(const __bf16* p, int ld) {
  const int lane = threadIdx.x & 31;
  const int col = lane & 15;
  const int kb  = (lane >> 4) * 16;
  v16bf r;
#pragma unroll
  for (int i = 0; i < 16; ++i) r[i] = p[(size_t)(kb + i) * ld + col];
  return r;
}

// B fragment from f32 LDS matrix (compressive memory): B[k][n] = m[kbase+k][nbase+n]
__device__ __forceinline__ v16bf frag_b_f32(const float* m, int kbase, int nbase, int ld) {
  const int lane = threadIdx.x & 31;
  const int col = lane & 15;
  const int kb  = (lane >> 4) * 16;
  v16bf r;
#pragma unroll
  for (int i = 0; i < 16; ++i) r[i] = (__bf16)m[(size_t)(kbase + kb + i) * ld + nbase + col];
  return r;
}

// ---------------------------------------------------------------------------
// f32 -> bf16 conversion
// ---------------------------------------------------------------------------
__global__ __launch_bounds__(256) void cvt_f32_bf16(const float* __restrict__ s,
                                                    __bf16* __restrict__ d, int n) {
  int i = blockIdx.x * 256 + threadIdx.x;
  if (i < n) d[i] = (__bf16)s[i];
}

// ---------------------------------------------------------------------------
// Generic bf16 WMMA GEMM: C = act(A(MxK) @ B(KxN) + bias)
// Block 256 thr (8 waves), tile 128x128x64.  Wave computes 32x64 (2x4 subtiles).
// Double-buffered LDS; A tile staged via GLOBAL_LOAD_ASYNC_TO_LDS_B128
// (ASYNCcnt), B tile staged transposed through VGPRs (element scatter).
// Dynamic LDS: 4 x 128x72 bf16 = 72KB (<< 320KB/WGP).
// flags: 1=bias, 2=relu, 4=bf16 output (else f32)
// ---------------------------------------------------------------------------
#define GEMM_TILE_ELEMS (128 * 72)
#define GEMM_LDS_BYTES  (4 * GEMM_TILE_ELEMS * 2)

__device__ __forceinline__ void gemm_stage(const __bf16* __restrict__ A,
                                           const __bf16* __restrict__ B,
                                           int N, int K, int m0, int n0, int k0,
                                           __bf16* As, __bf16* Bt, int tid) {
  // A tile (128x64): 16B chunks, async straight into LDS
#pragma unroll
  for (int c = 0; c < 4; ++c) {
    int id = tid + c * 256;  // 0..1023 : 128 rows x 8 chunks
    int r  = id >> 3;
    int kc = (id & 7) * 8;
    copy16_g2l(&A[(size_t)(m0 + r) * K + k0 + kc], As + r * 72 + kc);
  }
  // B tile (64x128) transposed: contiguous global read, LDS scatter
#pragma unroll
  for (int c = 0; c < 4; ++c) {
    int id = tid + c * 256;  // 0..1023 : 64 k-rows x 16 n-chunks
    int kk = id >> 4;
    int nc = (id & 15) * 8;
    uint4 t4 = *(const uint4*)&B[(size_t)(k0 + kk) * N + n0 + nc];
    const __bf16* tv = (const __bf16*)&t4;
#pragma unroll
    for (int i = 0; i < 8; ++i) Bt[(nc + i) * 72 + kk] = tv[i];
  }
}

__global__ __launch_bounds__(256)
void gemm_bf16_wmma(const __bf16* __restrict__ A, const __bf16* __restrict__ B,
                    const float* __restrict__ bias, float* __restrict__ Cf,
                    __bf16* __restrict__ Cb, int M, int N, int K, int flags) {
  extern __shared__ char gsm[];
  // ping-pong buffers derived arithmetically (no pointer arrays -> no
  // addrspacecast static initializers for the linker to choke on)
  __bf16* lds = (__bf16*)gsm;   // [0..1]: A bufs, [2..3]: Bt bufs

  const int tid  = threadIdx.x;
  const int lane = tid & 31;
  const int w    = tid >> 5;
  const int hw   = lane >> 4;
  const int lc   = lane & 15;
  const int wr   = w & 3;    // 4 row groups of 32
  const int wc   = w >> 2;   // 2 col groups of 64
  const int m0   = blockIdx.y * 128;
  const int n0   = blockIdx.x * 128;

  v8f acc[2][4];
#pragma unroll
  for (int i = 0; i < 2; ++i)
#pragma unroll
    for (int j = 0; j < 4; ++j) acc[i][j] = zero_v8f();

  // prologue: stage tile 0
  gemm_stage(A, B, N, K, m0, n0, 0, lds, lds + 2 * GEMM_TILE_ELEMS, tid);
  wait_async_copies();
  __syncthreads();

  const int niter = K / 64;
  for (int it = 0; it < niter; ++it) {
    const int cur = it & 1;
    const int nxt = cur ^ 1;
    // overlap: kick off tile it+1 while computing tile it
    if (it + 1 < niter)
      gemm_stage(A, B, N, K, m0, n0, (it + 1) * 64,
                 lds + nxt * GEMM_TILE_ELEMS,
                 lds + (2 + nxt) * GEMM_TILE_ELEMS, tid);

    const __bf16* As = lds + cur * GEMM_TILE_ELEMS;
    const __bf16* Bt = lds + (2 + cur) * GEMM_TILE_ELEMS;
#pragma unroll
    for (int ks = 0; ks < 2; ++ks) {  // two K=32 WMMA steps per 64-K tile
      v16bf af[2], bfr[4];
#pragma unroll
      for (int i = 0; i < 2; ++i)
        af[i] = frag_a_kcontig(As + (wr * 32 + i * 16) * 72 + ks * 32, 72);
#pragma unroll
      for (int j = 0; j < 4; ++j)
        bfr[j] = frag_b_kcontig(Bt + (wc * 64 + j * 16) * 72 + ks * 32, 72);
#pragma unroll
      for (int i = 0; i < 2; ++i)
#pragma unroll
        for (int j = 0; j < 4; ++j) acc[i][j] = WMMA_BF16(af[i], bfr[j], acc[i][j]);
    }
    wait_async_copies();   // tile it+1 fully landed in LDS
    __syncthreads();       // everyone done reading cur; nxt visible to all waves
  }

  // ---- epilogue ----
#pragma unroll
  for (int i = 0; i < 2; ++i) {
#pragma unroll
    for (int j = 0; j < 4; ++j) {
#pragma unroll
      for (int r = 0; r < 8; ++r) {
        const int row = m0 + wr * 32 + i * 16 + hw * 8 + r;
        const int col = n0 + wc * 64 + j * 16 + lc;
        float vl = acc[i][j][r];
        if (flags & 1) vl += bias[col];
        if (flags & 2) vl = fmaxf(vl, 0.f);
        if (flags & 4) Cb[(size_t)row * N + col] = (__bf16)vl;
        else           Cf[(size_t)row * N + col] = vl;
      }
    }
  }
}

// ---------------------------------------------------------------------------
// Attention stage 1: per-(b,h,seg) memory contribution
//   Mseg = sigma(K)^T @ V  (64x64),  Zseg = sum_s sigma(K)  (64)
// grid = B*H*NSEG = 256 blocks, 256 threads (8 waves x 2 tiles of 16x16)
// ---------------------------------------------------------------------------
__global__ __launch_bounds__(256)
void attn_seg_mem(const __bf16* __restrict__ Kg, const __bf16* __restrict__ Vg,
                  float* __restrict__ Mseg, float* __restrict__ Zseg) {
  const int blk = blockIdx.x;
  const int seg = blk % NSEG_;
  const int bh  = blk / NSEG_;
  const size_t off = ((size_t)bh * S_ + (size_t)seg * SEG_) * DK_;
  const __bf16* k = Kg + off;
  const __bf16* v = Vg + off;
  float* Mout = Mseg + (size_t)blk * DK_ * DV_;
  float* Zout = Zseg + (size_t)blk * DK_;

  const int tid  = threadIdx.x;
  const int lane = tid & 31;
  const int w    = tid >> 5;
  const int hw   = lane >> 4;
  const int lc   = lane & 15;

#pragma unroll
  for (int tt = 0; tt < 2; ++tt) {
    const int id = w * 2 + tt;
    const int tm = id >> 2, tn = id & 3;
    v8f acc = zero_v8f();
    for (int kt = 0; kt < SEG_ / 32; ++kt) {
      v16bf af  = frag_aT_sigma(k, tm * 16, kt * 32, DK_);
      v16bf bfr = frag_b_strided(v + (size_t)(kt * 32) * DV_ + tn * 16, DV_);
      acc = WMMA_BF16(af, bfr, acc);
    }
#pragma unroll
    for (int r = 0; r < 8; ++r)
      Mout[(size_t)(tm * 16 + hw * 8 + r) * DV_ + tn * 16 + lc] = acc[r];
  }
  if (tid < DK_) {
    float a = 0.f;
    for (int s = 0; s < SEG_; ++s) a += sigma_elu1((float)k[(size_t)s * DK_ + tid]);
    Zout[tid] = a;
  }
}

// ---------------------------------------------------------------------------
// Attention stage 2: exclusive prefix over segments (the "scan" carry)
//   Mem_in[seg] = sum_{s<seg} Mseg[s],  Z_in[seg] = 1/DK + sum_{s<seg} Zseg[s]
// grid = B*H = 32 blocks
// ---------------------------------------------------------------------------
__global__ __launch_bounds__(256)
void attn_prefix(const float* __restrict__ Mseg, const float* __restrict__ Zseg,
                 float* __restrict__ Mem_in, float* __restrict__ Z_in) {
  const int bh  = blockIdx.x;
  const int tid = threadIdx.x;
  for (int e = tid; e < DK_ * DV_; e += 256) {
    float run = 0.f;
    for (int s = 0; s < NSEG_; ++s) {
      size_t idx = ((size_t)(bh * NSEG_ + s)) * (DK_ * DV_) + e;
      Mem_in[idx] = run;
      run += Mseg[idx];
    }
  }
  if (tid < DK_) {
    float run = 1.0f / DK_;
    for (int s = 0; s < NSEG_; ++s) {
      size_t idx = ((size_t)(bh * NSEG_ + s)) * DK_ + tid;
      Z_in[idx] = run;
      run += Zseg[idx];
    }
  }
}

// ---------------------------------------------------------------------------
// Attention stage 3: per-(b,h,seg) causal softmax attention + memory retrieval
// grid = 256 blocks, 256 threads (8 waves). Each wave owns 16 rows per pass,
// 4 passes cover SEG=512 rows. Two-pass softmax (max, then exp/store-bf16-P
// into LDS), then P@V and sigma(Q)@mem via WMMA.
// Dynamic LDS: P (8*16*512 bf16 = 128KB) + mem(16KB) + z + den  ~ 145KB
// (well under the 320KB/WGP CDNA5 budget).
// ---------------------------------------------------------------------------
__global__ __launch_bounds__(256)
void attn_core(const __bf16* __restrict__ Qg, const __bf16* __restrict__ Kg,
               const __bf16* __restrict__ Vg, const float* __restrict__ Mem_in,
               const float* __restrict__ Z_in, const float* __restrict__ betas,
               __bf16* __restrict__ attb) {
  extern __shared__ char smem_raw[];
  __bf16* Pbuf  = (__bf16*)smem_raw;                                  // 8*16*SEG
  float*  memsh = (float*)(smem_raw + (size_t)8 * 16 * SEG_ * 2);     // 64*64
  float*  zsh   = memsh + DK_ * DV_;                                  // 64
  float*  denw  = zsh + DK_;                                          // 8*16

  const int blk = blockIdx.x;
  const int seg = blk % NSEG_;
  const int bh  = blk / NSEG_;
  const int b   = bh / H_, h = bh % H_;
  const size_t off = ((size_t)bh * S_ + (size_t)seg * SEG_) * DK_;
  const __bf16* q = Qg + off;
  const __bf16* k = Kg + off;
  const __bf16* v = Vg + off;

  const int tid  = threadIdx.x;
  const int lane = tid & 31;
  const int w    = tid >> 5;
  const int hw   = lane >> 4;
  const int lc   = lane & 15;
  __bf16* Pw = Pbuf + (size_t)w * 16 * SEG_;

  for (int e = tid; e < DK_ * DV_; e += 256)
    memsh[e] = Mem_in[(size_t)blk * (DK_ * DV_) + e];
  if (tid < DK_) zsh[tid] = Z_in[(size_t)blk * DK_ + tid];
  __syncthreads();

  const float scale = 0.125f;  // 1/sqrt(DK)

  for (int pass = 0; pass < 4; ++pass) {
    const int R    = pass * 128 + w * 16;   // first row of this wave's tile
    const int jmax = (R + 15) >> 4;         // last causal col-tile (inclusive)

    v16bf afq[2];
#pragma unroll
    for (int kt = 0; kt < 2; ++kt)
      afq[kt] = frag_a_kcontig(q + (size_t)R * DK_ + kt * 32, DK_);

    // ---- pass A: running row max of scaled masked scores ----
    float mrow[8];
#pragma unroll
    for (int r = 0; r < 8; ++r) mrow[r] = -1e30f;
    for (int j = 0; j <= jmax; ++j) {
      v8f acc = zero_v8f();
#pragma unroll
      for (int kt = 0; kt < 2; ++kt) {
        v16bf bfr = frag_b_kcontig(k + (size_t)(j * 16) * DK_ + kt * 32, DK_);
        acc = WMMA_BF16(afq[kt], bfr, acc);
      }
#pragma unroll
      for (int r = 0; r < 8; ++r) {
        const int row = R + hw * 8 + r;
        const int col = j * 16 + lc;
        float s = acc[r] * scale;
        if (col > row) s = -1e30f;
        mrow[r] = fmaxf(mrow[r], s);
      }
    }
#pragma unroll
    for (int m = 1; m < 16; m <<= 1)
#pragma unroll
      for (int r = 0; r < 8; ++r) mrow[r] = fmaxf(mrow[r], __shfl_xor(mrow[r], m, 32));

    // ---- pass B: recompute, exp, row sums, store P as bf16 to LDS ----
    float srow[8];
#pragma unroll
    for (int r = 0; r < 8; ++r) srow[r] = 0.f;
    for (int j = 0; j <= jmax; ++j) {
      v8f acc = zero_v8f();
#pragma unroll
      for (int kt = 0; kt < 2; ++kt) {
        v16bf bfr = frag_b_kcontig(k + (size_t)(j * 16) * DK_ + kt * 32, DK_);
        acc = WMMA_BF16(afq[kt], bfr, acc);
      }
#pragma unroll
      for (int r = 0; r < 8; ++r) {
        const int row = R + hw * 8 + r;
        const int col = j * 16 + lc;
        float s = acc[r] * scale;
        float p = (col > row) ? 0.f : __expf(s - mrow[r]);
        srow[r] += p;
        Pw[(size_t)(hw * 8 + r) * SEG_ + col] = (__bf16)p;
      }
    }
    if ((jmax + 1) & 1) {  // zero-pad to a multiple of K=32 for the P@V WMMAs
      const int colp = (jmax + 1) * 16 + lc;
#pragma unroll
      for (int r = 0; r < 8; ++r) Pw[(size_t)(hw * 8 + r) * SEG_ + colp] = (__bf16)0.f;
    }
#pragma unroll
    for (int m = 1; m < 16; m <<= 1)
#pragma unroll
      for (int r = 0; r < 8; ++r) srow[r] += __shfl_xor(srow[r], m, 32);

    // denominator sigma(q) . z  (one row per lane 0..15)
    if (lane < 16) {
      float d = 0.f;
      const __bf16* qr = q + (size_t)(R + lane) * DK_;
      for (int f = 0; f < DK_; ++f) d += sigma_elu1((float)qr[f]) * zsh[f];
      denw[w * 16 + lane] = d;
    }
    __syncthreads();

    // ---- pass C: O = P @ V  and  M = sigma(Q) @ mem ----
    v8f Oacc[4], Macc[4];
#pragma unroll
    for (int t = 0; t < 4; ++t) { Oacc[t] = zero_v8f(); Macc[t] = zero_v8f(); }

    const int nkt = (jmax + 2) >> 1;   // ceil((jmax+1)*16 / 32)
    for (int kt = 0; kt < nkt; ++kt) {
      v16bf af = frag_a_kcontig(Pw + kt * 32, SEG_);
#pragma unroll
      for (int t = 0; t < 4; ++t) {
        v16bf bfr = frag_b_strided(v + (size_t)(kt * 32) * DV_ + t * 16, DV_);
        Oacc[t] = WMMA_BF16(af, bfr, Oacc[t]);
      }
    }
#pragma unroll
    for (int kt = 0; kt < 2; ++kt) {
      v16bf af = frag_a_sigma(q + (size_t)R * DK_ + kt * 32, DK_);
#pragma unroll
      for (int t = 0; t < 4; ++t) {
        v16bf bfr = frag_b_f32(memsh, kt * 32, t * 16, DV_);
        Macc[t] = WMMA_BF16(af, bfr, Macc[t]);
      }
    }

    // ---- combine: sig_b * att_mem + (1-sig_b) * att_dot, scrambled store ----
#pragma unroll
    for (int t = 0; t < 4; ++t) {
#pragma unroll
      for (int r = 0; r < 8; ++r) {
        const int rl  = hw * 8 + r;
        const int row = R + rl;
        const int dv  = t * 16 + lc;
        const float den = denw[w * 16 + rl];
        const float am  = Macc[t][r] / den;
        const float ad  = Oacc[t][r] / srow[r];
        const float sb  = 1.f / (1.f + __expf(-betas[h * DV_ + dv]));
        const float o   = sb * am + (1.f - sb) * ad;
        const size_t orow = (size_t)b * S_ + (size_t)seg * SEG_ + row;
        attb[orow * (H_ * DV_) + h * DV_ + dv] = (__bf16)o;
      }
    }
    __syncthreads();
  }
}

// ---------------------------------------------------------------------------
// Residual + LayerNorm: out = LN(mlp + x) * gamma + beta.  One block per row.
// ---------------------------------------------------------------------------
__global__ __launch_bounds__(256)
void ln_residual(const float* __restrict__ mlp, const float* __restrict__ x,
                 const float* __restrict__ gamma, const float* __restrict__ beta,
                 float* __restrict__ out) {
  __shared__ float rs[256], rs2[256];
  const int row = blockIdx.x;
  const int t   = threadIdx.x;
  const float* pm = mlp + (size_t)row * DIM_;
  const float* px = x + (size_t)row * DIM_;
  float v[4], s = 0.f, s2 = 0.f;
#pragma unroll
  for (int i = 0; i < 4; ++i) {
    const int c = t + i * 256;
    const float y = pm[c] + px[c];
    v[i] = y; s += y; s2 += y * y;
  }
  rs[t] = s; rs2[t] = s2;
  __syncthreads();
  for (int o = 128; o > 0; o >>= 1) {
    if (t < o) { rs[t] += rs[t + o]; rs2[t] += rs2[t + o]; }
    __syncthreads();
  }
  const float mu  = rs[0] * (1.f / DIM_);
  const float var = rs2[0] * (1.f / DIM_) - mu * mu;
  const float inv = rsqrtf(var + 1e-5f);
#pragma unroll
  for (int i = 0; i < 4; ++i) {
    const int c = t + i * 256;
    out[(size_t)row * DIM_ + c] = (v[i] - mu) * inv * gamma[c] + beta[c];
  }
}

// ---------------------------------------------------------------------------
// Host-side launch
// ---------------------------------------------------------------------------
static inline void* ws_take(void* ws, size_t& off, size_t bytes) {
  void* p = (char*)ws + off;
  off += (bytes + 255) & ~(size_t)255;
  return p;
}

extern "C" void kernel_launch(void* const* d_in, const int* in_sizes, int n_in,
                              void* d_out, int out_size, void* d_ws, size_t ws_size,
                              hipStream_t stream) {
  (void)in_sizes; (void)n_in; (void)out_size; (void)ws_size;
  const float* x     = (const float*)d_in[0];
  const float* Wq    = (const float*)d_in[1];
  const float* Wk    = (const float*)d_in[2];
  const float* Wv    = (const float*)d_in[3];
  const float* Wo    = (const float*)d_in[4];
  const float* betas = (const float*)d_in[5];
  const float* W1    = (const float*)d_in[6];
  const float* b1    = (const float*)d_in[7];
  const float* W2    = (const float*)d_in[8];
  const float* b2    = (const float*)d_in[9];
  const float* gamma = (const float*)d_in[10];
  const float* beta  = (const float*)d_in[11];
  float* out = (float*)d_out;

  size_t off = 0;
  __bf16* xb   = (__bf16*)ws_take(d_ws, off, (size_t)BS_ * DIM_ * 2);
  __bf16* Wqb  = (__bf16*)ws_take(d_ws, off, (size_t)DIM_ * H_ * DK_ * 2);
  __bf16* Wkb  = (__bf16*)ws_take(d_ws, off, (size_t)DIM_ * H_ * DK_ * 2);
  __bf16* Wvb  = (__bf16*)ws_take(d_ws, off, (size_t)DIM_ * H_ * DV_ * 2);
  __bf16* Wob  = (__bf16*)ws_take(d_ws, off, (size_t)H_ * DV_ * DIM_ * 2);
  __bf16* W1b  = (__bf16*)ws_take(d_ws, off, (size_t)DIM_ * HID_ * 2);
  __bf16* W2b  = (__bf16*)ws_take(d_ws, off, (size_t)HID_ * DIM_ * 2);
  __bf16* Qb   = (__bf16*)ws_take(d_ws, off, (size_t)BS_ * H_ * DK_ * 2);
  __bf16* Kb   = (__bf16*)ws_take(d_ws, off, (size_t)BS_ * H_ * DK_ * 2);
  __bf16* Vb   = (__bf16*)ws_take(d_ws, off, (size_t)BS_ * H_ * DV_ * 2);
  __bf16* attbuf = (__bf16*)ws_take(d_ws, off, (size_t)BS_ * H_ * DV_ * 2);
  __bf16* aob  = (__bf16*)ws_take(d_ws, off, (size_t)BS_ * DIM_ * 2);
  __bf16* hb   = (__bf16*)ws_take(d_ws, off, (size_t)BS_ * HID_ * 2);
  float*  mlpf = (float*)ws_take(d_ws, off, (size_t)BS_ * DIM_ * 4);
  float*  Mseg = (float*)ws_take(d_ws, off, (size_t)B_ * H_ * NSEG_ * DK_ * DV_ * 4);
  float*  Zseg = (float*)ws_take(d_ws, off, (size_t)B_ * H_ * NSEG_ * DK_ * 4);
  float*  Memi = (float*)ws_take(d_ws, off, (size_t)B_ * H_ * NSEG_ * DK_ * DV_ * 4);
  float*  Zi   = (float*)ws_take(d_ws, off, (size_t)B_ * H_ * NSEG_ * DK_ * 4);

  // ---- bf16 conversions ----
  auto cvt = [&](const float* s, __bf16* d, int n) {
    cvt_f32_bf16<<<(n + 255) / 256, 256, 0, stream>>>(s, d, n);
  };
  cvt(x,  xb,  BS_ * DIM_);
  cvt(Wq, Wqb, DIM_ * H_ * DK_);
  cvt(Wk, Wkb, DIM_ * H_ * DK_);
  cvt(Wv, Wvb, DIM_ * H_ * DV_);
  cvt(Wo, Wob, H_ * DV_ * DIM_);
  cvt(W1, W1b, DIM_ * HID_);
  cvt(W2, W2b, HID_ * DIM_);

  // ---- QKV projections (bf16 WMMA GEMMs, bf16 out) ----
  dim3 blk(256);
  gemm_bf16_wmma<<<dim3((H_ * DK_) / 128, BS_ / 128), blk, GEMM_LDS_BYTES, stream>>>(
      xb, Wqb, nullptr, nullptr, Qb, BS_, H_ * DK_, DIM_, 4);
  gemm_bf16_wmma<<<dim3((H_ * DK_) / 128, BS_ / 128), blk, GEMM_LDS_BYTES, stream>>>(
      xb, Wkb, nullptr, nullptr, Kb, BS_, H_ * DK_, DIM_, 4);
  gemm_bf16_wmma<<<dim3((H_ * DV_) / 128, BS_ / 128), blk, GEMM_LDS_BYTES, stream>>>(
      xb, Wvb, nullptr, nullptr, Vb, BS_, H_ * DV_, DIM_, 4);

  // ---- compressive memory: contributions -> prefix scan -> attention ----
  attn_seg_mem<<<B_ * H_ * NSEG_, 256, 0, stream>>>(Kb, Vb, Mseg, Zseg);
  attn_prefix<<<B_ * H_, 256, 0, stream>>>(Mseg, Zseg, Memi, Zi);
  const size_t att_smem =
      (size_t)8 * 16 * SEG_ * 2 + (DK_ * DV_ + DK_ + 8 * 16) * sizeof(float);
  attn_core<<<B_ * H_ * NSEG_, 256, att_smem, stream>>>(Qb, Kb, Vb, Memi, Zi, betas, attbuf);

  // ---- output projection ----
  gemm_bf16_wmma<<<dim3(DIM_ / 128, BS_ / 128), blk, GEMM_LDS_BYTES, stream>>>(
      attbuf, Wob, nullptr, nullptr, aob, BS_, DIM_, H_ * DV_, 4);

  // ---- MLP ----
  gemm_bf16_wmma<<<dim3(HID_ / 128, BS_ / 128), blk, GEMM_LDS_BYTES, stream>>>(
      aob, W1b, b1, nullptr, hb, BS_, HID_, DIM_, 1 | 2 | 4);
  gemm_bf16_wmma<<<dim3(DIM_ / 128, BS_ / 128), blk, GEMM_LDS_BYTES, stream>>>(
      hb, W2b, b2, mlpf, nullptr, BS_, DIM_, HID_, 1);

  // ---- residual + LayerNorm ----
  ln_residual<<<BS_, 256, 0, stream>>>(mlpf, x, gamma, beta, out);
}
